// simBNN_7997229105510
// MI455X (gfx1250) — 16-bit and 8-bit matrix operand order repaired and checked by reading, compiled, not run on hardware
//
#include <hip/hip_runtime.h>
#include <hip/hip_bf16.h>
#include <stdint.h>


typedef __attribute__((ext_vector_type(8)))  int    v8i;
typedef __attribute__((ext_vector_type(16))) int    v16i;
typedef __attribute__((ext_vector_type(8)))  float  v8f;
typedef __attribute__((ext_vector_type(16))) __bf16 v16bf;

#define TPB 256
#define EPSBN 2e-5f

__device__ __forceinline__ unsigned char fsign_fp8(float f) {
    return (f > 0.f) ? (unsigned char)0x38
                     : ((f < 0.f) ? (unsigned char)0xB8 : (unsigned char)0x00);
}

__global__ void conv1_fp32(const float* __restrict__ x, const float* __restrict__ w,
                           const float* __restrict__ bias, float* __restrict__ y) {
    const int OH = 51, OW = 51, CO = 96, CI = 3, KH = 11, KW = 11, IH = 211, IW = 211;
    int idx = blockIdx.x * blockDim.x + threadIdx.x;
    int total = 8 * CO * OH * OW;
    if (idx >= total) return;
    int ow = idx % OW; int t = idx / OW;
    int oh = t % OH;   t /= OH;
    int co = t % CO;   int b = t / CO;
    float acc = bias[co];
    const float* xb = x + (size_t)b * CI * IH * IW;
    const float* wc = w + (size_t)co * CI * KH * KW;
    for (int ci = 0; ci < CI; ++ci)
        for (int kh = 0; kh < KH; ++kh) {
            const float* xr = xb + ((size_t)ci * IH + oh * 4 + kh) * IW + ow * 4;
            const float* wr = wc + (ci * KH + kh) * KW;
            #pragma unroll
            for (int kw = 0; kw < KW; ++kw) acc += xr[kw] * wr[kw];
        }
    y[idx] = acc;
}

__global__ void bn_stats(const float* __restrict__ x, float* __restrict__ mv,
                         int N, int C, int HW) {
    int c = blockIdx.x;
    float s = 0.f, s2 = 0.f;
    int per = N * HW;
    for (int i = threadIdx.x; i < per; i += TPB) {
        int n = i / HW, p = i - n * HW;
        float v = x[((size_t)n * C + c) * HW + p];
        s += v; s2 += v * v;
    }
    __shared__ float sh[2 * TPB];
    sh[threadIdx.x] = s; sh[TPB + threadIdx.x] = s2;
    __syncthreads();
    for (int o = TPB / 2; o > 0; o >>= 1) {
        if ((int)threadIdx.x < o) {
            sh[threadIdx.x]       += sh[threadIdx.x + o];
            sh[TPB + threadIdx.x] += sh[TPB + threadIdx.x + o];
        }
        __syncthreads();
    }
    if (threadIdx.x == 0) {
        float cnt = (float)per;
        float m = sh[0] / cnt;
        mv[c] = m;
        mv[C + c] = sh[TPB] / cnt - m * m;
    }
}

__global__ void bn_apply(float* __restrict__ x, const float* __restrict__ mv,
                         const float* __restrict__ g, const float* __restrict__ b,
                         int C, int HW, int total, int relu) {
    int idx = blockIdx.x * blockDim.x + threadIdx.x;
    if (idx >= total) return;
    int c = (idx / HW) % C;
    float y = g[c] * (x[idx] - mv[c]) * rsqrtf(mv[C + c] + EPSBN) + b[c];
    if (relu) y = fmaxf(y, 0.f);
    x[idx] = y;
}

__global__ void maxpool3(const float* __restrict__ x, float* __restrict__ y,
                         int NC, int H, int W, int OH, int OW) {
    int idx = blockIdx.x * blockDim.x + threadIdx.x;
    int total = NC * OH * OW;
    if (idx >= total) return;
    int ow = idx % OW; int t = idx / OW;
    int oh = t % OH;   int nc = t / OH;
    const float* xp = x + (size_t)nc * H * W;
    float m = -3.4e38f;
    #pragma unroll
    for (int kh = 0; kh < 3; ++kh)
        #pragma unroll
        for (int kw = 0; kw < 3; ++kw)
            m = fmaxf(m, xp[(oh * 2 + kh) * W + (ow * 2 + kw)]);
    y[idx] = m;
}

__global__ void im2col_sign(const float* __restrict__ x, unsigned char* __restrict__ out,
                            int C, int H, int W, int OH, int OW,
                            int KH, int KW, int pad, int K, int Kpad, int total) {
    int idx = blockIdx.x * blockDim.x + threadIdx.x;
    if (idx >= total) return;
    int k = idx % Kpad;
    int row = idx / Kpad;
    unsigned char val = 0;
    if (k < K) {
        int kw = k % KW; int t = k / KW;
        int kh = t % KH; int ci = t / KH;
        int ow = row % OW; int t2 = row / OW;
        int oh = t2 % OH;  int b = t2 / OH;
        int ih = oh - pad + kh;
        int iw = ow - pad + kw;
        if (ih >= 0 && ih < H && iw >= 0 && iw < W)
            val = fsign_fp8(x[(((size_t)b * C + ci) * H + ih) * W + iw]);
    }
    out[idx] = val;
}

__global__ void pack_sign_rows(const float* __restrict__ x, unsigned char* __restrict__ out,
                               int M, int K, int Kpad, int total) {
    int idx = blockIdx.x * blockDim.x + threadIdx.x;
    if (idx >= total) return;
    int k = idx % Kpad;
    int row = idx / Kpad;
    unsigned char v = 0;
    if (row < M && k < K) v = fsign_fp8(x[(size_t)row * K + k]);
    out[idx] = v;
}

__global__ void pack_bf16_rows(const float* __restrict__ x, unsigned short* __restrict__ out,
                               int M, int K, int Kpad, int total) {
    int idx = blockIdx.x * blockDim.x + threadIdx.x;
    if (idx >= total) return;
    int k = idx % Kpad;
    int row = idx / Kpad;
    float f = (row < M && k < K) ? x[(size_t)row * K + k] : 0.f;
    unsigned u = __float_as_uint(f);
    out[idx] = (unsigned short)((u + 0x7fffu + ((u >> 16) & 1u)) >> 16);
}

__device__ __forceinline__ v16i load_a_fp8(const unsigned char* Arow, int k0, int h) {
    v16i a;
    #pragma unroll
    for (int t = 0; t < 2; ++t)
        #pragma unroll
        for (int p = 0; p < 4; ++p) {
            int2 u = *(const int2*)(Arow + k0 + t * 64 + p * 16 + h * 8);
            a[8 * t + 2 * p] = u.x; a[8 * t + 2 * p + 1] = u.y;
        }
    return a;
}

__global__ void gemm_fp8(const unsigned char* __restrict__ A, const unsigned char* __restrict__ Bw,
                         float* __restrict__ out, const float* __restrict__ bias,
                         int M, int N, int Kpad, int scatterOHW) {
    const int lane = threadIdx.x & 31;
    const int wave = threadIdx.x >> 5;
    const int mT0 = blockIdx.x * 4;
    const int nT = blockIdx.y * 8 + wave;
    const int h = lane >> 4;
    const int l16 = lane & 15;
    const int ncol = nT * 16 + l16;
    const unsigned char* Brow = Bw + (size_t)ncol * Kpad;

    int r0 = (mT0 + 0) * 16 + l16; r0 = (r0 < M) ? r0 : (M - 1);
    int r1 = (mT0 + 1) * 16 + l16; r1 = (r1 < M) ? r1 : (M - 1);
    int r2 = (mT0 + 2) * 16 + l16; r2 = (r2 < M) ? r2 : (M - 1);
    int r3 = (mT0 + 3) * 16 + l16; r3 = (r3 < M) ? r3 : (M - 1);
    const unsigned char* A0 = A + (size_t)r0 * Kpad;
    const unsigned char* A1 = A + (size_t)r1 * Kpad;
    const unsigned char* A2 = A + (size_t)r2 * Kpad;
    const unsigned char* A3 = A + (size_t)r3 * Kpad;

    v8f acc0 = {}, acc1 = {}, acc2 = {}, acc3 = {};
    for (int k0 = 0; k0 < Kpad; k0 += 128) {
        v16i b = load_a_fp8(Brow, k0, h);
        v16i a0 = load_a_fp8(A0, k0, h);
        v16i a1 = load_a_fp8(A1, k0, h);
        v16i a2 = load_a_fp8(A2, k0, h);
        v16i a3 = load_a_fp8(A3, k0, h);
        acc0 = __builtin_amdgcn_wmma_f32_16x16x128_fp8_fp8(a0, b, (short)0, acc0, false, false);
        acc1 = __builtin_amdgcn_wmma_f32_16x16x128_fp8_fp8(a1, b, (short)0, acc1, false, false);
        acc2 = __builtin_amdgcn_wmma_f32_16x16x128_fp8_fp8(a2, b, (short)0, acc2, false, false);
        acc3 = __builtin_amdgcn_wmma_f32_16x16x128_fp8_fp8(a3, b, (short)0, acc3, false, false);
    }

    float bv = (bias && ncol < N) ? bias[ncol] : 0.f;
    #pragma unroll
    for (int i = 0; i < 4; ++i) {
        const v8f& acc = (i == 0) ? acc0 : (i == 1) ? acc1 : (i == 2) ? acc2 : acc3;
        #pragma unroll
        for (int r = 0; r < 8; ++r) {
            int m = (mT0 + i) * 16 + (h << 3) + r;
            if (m < M && ncol < N) {
                float val = acc[r] + bv;
                if (scatterOHW > 0) {
                    int bI = m / scatterOHW, s = m - bI * scatterOHW;
                    out[((size_t)bI * N + ncol) * scatterOHW + s] = val;
                } else {
                    out[(size_t)m * N + ncol] = val;
                }
            }
        }
    }
}

__global__ void gemm_bf16(const unsigned short* __restrict__ A, const unsigned short* __restrict__ Bw,
                          float* __restrict__ out, const float* __restrict__ bias,
                          int M, int Nout, int Kpad) {
    const int lane = threadIdx.x & 31;
    const int wave = threadIdx.x >> 5;
    const int mT = blockIdx.x;
    const int nT = blockIdx.y * 8 + wave;
    const int h = lane >> 4;
    const int l16 = lane & 15;
    const int row = mT * 16 + l16;
    const int rowC = (row < M) ? row : (M - 1);
    const int ncol = nT * 16 + l16;
    const unsigned short* Arow = A + (size_t)rowC * Kpad;
    const unsigned short* Brow = Bw + (size_t)ncol * Kpad;

    union Frag { v16bf v; v8i i; };
    v8f acc0 = {}, acc1 = {};
    for (int k0 = 0; k0 < Kpad; k0 += 64) {
        Frag a0, a1, b0, b1;
        #pragma unroll
        for (int p = 0; p < 2; ++p) {
            int4 t0 = *(const int4*)(Arow + k0 + p * 16 + h * 8);
            a0.i[4 * p] = t0.x; a0.i[4 * p + 1] = t0.y; a0.i[4 * p + 2] = t0.z; a0.i[4 * p + 3] = t0.w;
            int4 t1 = *(const int4*)(Arow + k0 + 32 + p * 16 + h * 8);
            a1.i[4 * p] = t1.x; a1.i[4 * p + 1] = t1.y; a1.i[4 * p + 2] = t1.z; a1.i[4 * p + 3] = t1.w;
        }
        #pragma unroll
        for (int p = 0; p < 2; ++p) {
            int4 t0 = *(const int4*)(Brow + k0 + p * 16 + h * 8);
            b0.i[4 * p] = t0.x; b0.i[4 * p + 1] = t0.y; b0.i[4 * p + 2] = t0.z; b0.i[4 * p + 3] = t0.w;
            int4 t1 = *(const int4*)(Brow + k0 + 32 + p * 16 + h * 8);
            b1.i[4 * p] = t1.x; b1.i[4 * p + 1] = t1.y; b1.i[4 * p + 2] = t1.z; b1.i[4 * p + 3] = t1.w;
        }
        acc0 = __builtin_amdgcn_wmma_f32_16x16x32_bf16(false, a0.v, false, b0.v,
                                                       (short)0, acc0, false, false);
        acc1 = __builtin_amdgcn_wmma_f32_16x16x32_bf16(false, a1.v, false, b1.v,
                                                       (short)0, acc1, false, false);
    }
    #pragma unroll
    for (int r = 0; r < 8; ++r) {
        int m = mT * 16 + (h << 3) + r;
        if (m < M && ncol < Nout)
            out[(size_t)m * Nout + ncol] = (acc0[r] + acc1[r]) + (bias ? bias[ncol] : 0.f);
    }
}

__global__ void softmax_ch(const float* __restrict__ lg, float* __restrict__ out) {
    const int TOT = 8 * 5 * 1000;
    int idx = blockIdx.x * blockDim.x + threadIdx.x;
    if (idx >= TOT) return;
    int n = idx % 1000; int t = idx / 1000;
    int hh = t % 5;     int b = t / 5;
    size_t base = ((size_t)b * 1280 + hh) * 1000 + n;
    float mx = -3.4e38f;
    for (int c = 0; c < 256; ++c) mx = fmaxf(mx, lg[base + (size_t)c * 5000]);
    float s = 0.f;
    for (int c = 0; c < 256; ++c) s += expf(lg[base + (size_t)c * 5000] - mx);
    float inv = 1.f / s;
    for (int c = 0; c < 256; ++c)
        out[base + (size_t)c * 5000] = expf(lg[base + (size_t)c * 5000] - mx) * inv;
}

extern "C" void kernel_launch(void* const* d_in, const int* in_sizes, int n_in,
                              void* d_out, int out_size, void* d_ws, size_t ws_size,
                              hipStream_t stream) {
    (void)in_sizes; (void)n_in; (void)out_size; (void)ws_size;
    const float* x       = (const float*)d_in[0];
    const float* conv1_w = (const float*)d_in[1];  const float* conv1_b = (const float*)d_in[2];
    const float* bn1_g   = (const float*)d_in[3];  const float* bn1_b   = (const float*)d_in[4];
    const float* conv2_w = (const float*)d_in[5];  const float* conv2_b = (const float*)d_in[6];
    const float* bn2_g   = (const float*)d_in[7];  const float* bn2_b   = (const float*)d_in[8];
    const float* conv3_w = (const float*)d_in[9];  const float* conv3_b = (const float*)d_in[10];
    const float* bn3_g   = (const float*)d_in[11]; const float* bn3_b   = (const float*)d_in[12];
    const float* conv4_w = (const float*)d_in[13]; const float* conv4_b = (const float*)d_in[14];
    const float* bn4_g   = (const float*)d_in[15]; const float* bn4_b   = (const float*)d_in[16];
    const float* conv5_w = (const float*)d_in[17]; const float* conv5_b = (const float*)d_in[18];
    const float* bn5_g   = (const float*)d_in[19]; const float* bn5_b   = (const float*)d_in[20];
    const float* fc1_w   = (const float*)d_in[21];
    const float* bn6_g   = (const float*)d_in[22]; const float* bn6_b   = (const float*)d_in[23];
    const float* fc2_w   = (const float*)d_in[24];
    const float* bn7_g   = (const float*)d_in[25]; const float* bn7_b   = (const float*)d_in[26];
    const float* fc3_w   = (const float*)d_in[27]; const float* fc3_b   = (const float*)d_in[28];

    char* ws = (char*)d_ws;
    size_t off = 0;
    auto alloc = [&](size_t bytes) -> void* {
        void* p = ws + off;
        off += (bytes + 255) & ~(size_t)255;
        return p;
    };

    float*          c1    = (float*)         alloc((size_t)1997568 * 4);
    float*          p1    = (float*)         alloc((size_t)480000  * 4);
    unsigned char*  col2  = (unsigned char*) alloc((size_t)5000 * 2560);
    unsigned char*  w2s   = (unsigned char*) alloc((size_t)256  * 2560);
    float*          c2    = (float*)         alloc((size_t)1280000 * 4);
    float*          p2    = (float*)         alloc((size_t)294912  * 4);
    unsigned char*  col3  = (unsigned char*) alloc((size_t)1152 * 2304);
    unsigned char*  w3s   = (unsigned char*) alloc((size_t)384  * 2304);
    float*          c3    = (float*)         alloc((size_t)442368 * 4);
    unsigned char*  col4  = (unsigned char*) alloc((size_t)1152 * 3584);
    unsigned char*  w4s   = (unsigned char*) alloc((size_t)384  * 3584);
    float*          c4    = (float*)         alloc((size_t)442368 * 4);
    unsigned char*  col5  = (unsigned char*) alloc((size_t)1152 * 3584);
    unsigned char*  w5s   = (unsigned char*) alloc((size_t)256  * 3584);
    float*          c5    = (float*)         alloc((size_t)294912 * 4);
    float*          p5    = (float*)         alloc((size_t)51200  * 4);
    unsigned char*  a1    = (unsigned char*) alloc((size_t)10240 * 256);
    unsigned char*  w1fs  = (unsigned char*) alloc((size_t)4096  * 256);
    float*          fcbuf = (float*)         alloc((size_t)10240 * 4096 * 4);
    unsigned char*  a2    = (unsigned char*) alloc((size_t)10240 * 4096);
    unsigned char*  w2fs  = (unsigned char*) alloc((size_t)4096  * 4096);
    unsigned short* a3bf  = (unsigned short*)alloc((size_t)10240 * 4096 * 2);
    unsigned short* w3bf  = (unsigned short*)alloc((size_t)1024  * 4096 * 2);
    float*          mv    = (float*)         alloc(4096);
    float*          fc3o  = (float*)a2;

    auto G = [](int n) { return dim3((unsigned)((n + TPB - 1) / TPB)); };

    conv1_fp32<<<G(1997568), TPB, 0, stream>>>(x, conv1_w, conv1_b, c1);
    bn_stats  <<<96,  TPB, 0, stream>>>(c1, mv, 8, 96, 2601);
    bn_apply  <<<G(1997568), TPB, 0, stream>>>(c1, mv, bn1_g, bn1_b, 96, 2601, 1997568, 1);
    maxpool3  <<<G(480000),  TPB, 0, stream>>>(c1, p1, 768, 51, 51, 25, 25);

    im2col_sign  <<<G(12800000), TPB, 0, stream>>>(p1, col2, 96, 25, 25, 25, 25, 5, 5, 2, 2400, 2560, 12800000);
    pack_sign_rows<<<G(655360),  TPB, 0, stream>>>(conv2_w, w2s, 256, 2400, 2560, 655360);
    gemm_fp8 <<<dim3(79, 2), TPB, 0, stream>>>(col2, w2s, c2, conv2_b, 5000, 256, 2560, 625);
    bn_stats <<<256, TPB, 0, stream>>>(c2, mv, 8, 256, 625);
    bn_apply <<<G(1280000), TPB, 0, stream>>>(c2, mv, bn2_g, bn2_b, 256, 625, 1280000, 0);
    maxpool3 <<<G(294912),  TPB, 0, stream>>>(c2, p2, 2048, 25, 25, 12, 12);

    im2col_sign  <<<G(2654208), TPB, 0, stream>>>(p2, col3, 256, 12, 12, 12, 12, 3, 3, 1, 2304, 2304, 2654208);
    pack_sign_rows<<<G(884736), TPB, 0, stream>>>(conv3_w, w3s, 384, 2304, 2304, 884736);
    gemm_fp8 <<<dim3(18, 3), TPB, 0, stream>>>(col3, w3s, c3, conv3_b, 1152, 384, 2304, 144);
    bn_stats <<<384, TPB, 0, stream>>>(c3, mv, 8, 384, 144);
    bn_apply <<<G(442368), TPB, 0, stream>>>(c3, mv, bn3_g, bn3_b, 384, 144, 442368, 0);

    im2col_sign  <<<G(4128768), TPB, 0, stream>>>(c3, col4, 384, 12, 12, 12, 12, 3, 3, 1, 3456, 3584, 4128768);
    pack_sign_rows<<<G(1376256), TPB, 0, stream>>>(conv4_w, w4s, 384, 3456, 3584, 1376256);
    gemm_fp8 <<<dim3(18, 3), TPB, 0, stream>>>(col4, w4s, c4, conv4_b, 1152, 384, 3584, 144);
    bn_stats <<<384, TPB, 0, stream>>>(c4, mv, 8, 384, 144);
    bn_apply <<<G(442368), TPB, 0, stream>>>(c4, mv, bn4_g, bn4_b, 384, 144, 442368, 0);

    im2col_sign  <<<G(4128768), TPB, 0, stream>>>(c4, col5, 384, 12, 12, 12, 12, 3, 3, 1, 3456, 3584, 4128768);
    pack_sign_rows<<<G(917504), TPB, 0, stream>>>(conv5_w, w5s, 256, 3456, 3584, 917504);
    gemm_fp8 <<<dim3(18, 2), TPB, 0, stream>>>(col5, w5s, c5, conv5_b, 1152, 256, 3584, 144);
    bn_stats <<<256, TPB, 0, stream>>>(c5, mv, 8, 256, 144);
    bn_apply <<<G(294912), TPB, 0, stream>>>(c5, mv, bn5_g, bn5_b, 256, 144, 294912, 0);
    maxpool3 <<<G(51200), TPB, 0, stream>>>(c5, p5, 2048, 12, 12, 5, 5);

    pack_sign_rows<<<G(2621440), TPB, 0, stream>>>(p5, a1, 10240, 5, 256, 2621440);
    pack_sign_rows<<<G(1048576), TPB, 0, stream>>>(fc1_w, w1fs, 4096, 5, 256, 1048576);
    gemm_fp8 <<<dim3(160, 32), TPB, 0, stream>>>(a1, w1fs, fcbuf, nullptr, 10240, 4096, 256, 0);
    bn_stats <<<256, TPB, 0, stream>>>(fcbuf, mv, 8, 256, 20480);
    bn_apply <<<G(41943040), TPB, 0, stream>>>(fcbuf, mv, bn6_g, bn6_b, 256, 20480, 41943040, 1);

    pack_sign_rows<<<G(41943040), TPB, 0, stream>>>(fcbuf, a2, 10240, 4096, 4096, 41943040);
    pack_sign_rows<<<G(16777216), TPB, 0, stream>>>(fc2_w, w2fs, 4096, 4096, 4096, 16777216);
    gemm_fp8 <<<dim3(160, 32), TPB, 0, stream>>>(a2, w2fs, fcbuf, nullptr, 10240, 4096, 4096, 0);
    bn_stats <<<256, TPB, 0, stream>>>(fcbuf, mv, 8, 256, 20480);
    bn_apply <<<G(41943040), TPB, 0, stream>>>(fcbuf, mv, bn7_g, bn7_b, 256, 20480, 41943040, 1);

    pack_bf16_rows<<<G(41943040), TPB, 0, stream>>>(fcbuf, a3bf, 10240, 4096, 4096, 41943040);
    pack_bf16_rows<<<G(4194304),  TPB, 0, stream>>>(fc3_w, w3bf, 1000, 4096, 4096, 4194304);
    gemm_bf16 <<<dim3(640, 8), TPB, 0, stream>>>(a3bf, w3bf, fc3o, fc3_b, 10240, 1000, 4096);
    softmax_ch<<<G(40000), TPB, 0, stream>>>(fc3o, (float*)d_out);
}
